// DeepFieldWeightedFactorizationMachineModel_14147622273357
// MI455X (gfx1250) — compile-verified, matrix operand order checked by reading, not run on hardware
//
#include <hip/hip_runtime.h>
#include <hip/hip_bf16.h>
#include <math.h>

// ---------------- constants ----------------
#define NUM_FIELDS 39
#define FIELD_SIZE 2560
#define VOCAB      (NUM_FIELDS * FIELD_SIZE)   // 99840
#define EMBED_DIM  16
#define BATCH      4096
#define IN0        (NUM_FIELDS * EMBED_DIM)    // 624
#define K0P        640                          // 624 padded to mult of 32
#define HID        400
#define HP         416                          // 400 padded to mult of 32 (K and N pad)
#define BN_EPS     1e-5f

// ---------------- WMMA types ----------------
typedef __attribute__((ext_vector_type(16))) __bf16 v16bf;
typedef __attribute__((ext_vector_type(8)))  float  v8f;

union FragU { uint4 u[2]; v16bf v; };

__device__ __forceinline__ unsigned short f2bf(float f) {
  unsigned u = __builtin_bit_cast(unsigned, f);
  unsigned r = u + 0x7fffu + ((u >> 16) & 1u);   // round-to-nearest-even
  return (unsigned short)(r >> 16);
}
__device__ __forceinline__ float bf2f(unsigned short h) {
  unsigned u = ((unsigned)h) << 16;
  return __builtin_bit_cast(float, u);
}

// ---------------- weight transpose + bf16 convert: Wt[n][k] = bf16(W[k][n]) ----------------
// Wt is (Npad x Kpad); rows n >= N and cols k >= K are zero.
__global__ __launch_bounds__(256) void k_convert_wt(const float* __restrict__ W,
                                                    unsigned short* __restrict__ Wt,
                                                    int K, int N, int Kpad, int Npad) {
  int i = blockIdx.x * blockDim.x + threadIdx.x;
  int total = Npad * Kpad;
  if (i >= total) return;
  int n = i / Kpad, k = i - n * Kpad;
  Wt[(size_t)n * Kpad + k] = (k < K && n < N) ? f2bf(W[(size_t)k * N + n]) : (unsigned short)0;
}

// ---------------- embedding gather + FM second order (one wave per batch row) ----------------
__global__ __launch_bounds__(256) void k_gather_fm(const long long* __restrict__ x,
                                                   const float* __restrict__ emb,
                                                   const float* __restrict__ cov,
                                                   unsigned short* __restrict__ h0,
                                                   float* __restrict__ secsum) {
  __shared__ float e[8][IN0];
  int wid  = threadIdx.x >> 5;
  int lane = threadIdx.x & 31;
  int b = blockIdx.x * 8 + wid;

  const long long* xb = x + (size_t)b * NUM_FIELDS;
  for (int t = lane; t < IN0; t += 32) {
    int f = t >> 4, d = t & 15;
    long long idx = xb[f];
    size_t row = (size_t)f * VOCAB + (size_t)f * FIELD_SIZE + (size_t)idx;
    float v = emb[row * EMBED_DIM + d];
    e[wid][t] = v;
    h0[(size_t)b * K0P + t] = f2bf(v);
  }
  if (lane < K0P - IN0) h0[(size_t)b * K0P + IN0 + lane] = 0;
  __syncthreads();

  if (lane < EMBED_DIM) {
    int d = lane;
    float sfull = 0.f, sdiag = 0.f;
    for (int k = 0; k < NUM_FIELDS; ++k) {
      float ek = e[wid][k * EMBED_DIM + d];
      float tk = 0.f;
      for (int l = 0; l < NUM_FIELDS; ++l) {
        float cs = 0.5f * (cov[k * NUM_FIELDS + l] + cov[l * NUM_FIELDS + k]);
        tk += cs * e[wid][l * EMBED_DIM + d];
      }
      sfull += tk * ek;
      sdiag += cov[k * NUM_FIELDS + k] * ek * ek;
    }
    float s = 0.5f * (sfull - sdiag);
    s += __shfl_xor(s, 1, 32);
    s += __shfl_xor(s, 2, 32);
    s += __shfl_xor(s, 4, 32);
    s += __shfl_xor(s, 8, 32);
    if (d == 0) secsum[b] = s;
  }
}

// ---------------- bf16 WMMA GEMM + bias + BN(eval) + ReLU, 2M x 2N register blocking ----
// X: M x Kpad (bf16, row-major). Wt: Npad x Kpad (bf16, W transposed, zero-padded).
// Y: M x ldY (bf16). Each wave computes a 32x32 output tile = 4 WMMA accumulators,
// amortizing fragment loads (2 A + 2 B b128-pairs feed 4 v_wmma per K-step).
// Columns n >= Nvalid are forced to zero so K-padding of the next layer is pre-zeroed.
__global__ __launch_bounds__(256) void k_gemm_bn_relu(
    const unsigned short* __restrict__ X, const unsigned short* __restrict__ Wt,
    unsigned short* __restrict__ Y,
    const float* __restrict__ bias, const float* __restrict__ gam,
    const float* __restrict__ bet,  const float* __restrict__ rmean,
    const float* __restrict__ rvar,
    int mgroups, int ngroups, int KT, int Kpad, int ldY, int Nvalid) {
  int wave = blockIdx.x * (blockDim.x >> 5) + (threadIdx.x >> 5);
  if (wave >= mgroups * ngroups) return;          // wave-uniform: EXEC stays all-1s
  int mg = wave % mgroups, ng = wave / mgroups;
  int lane = threadIdx.x & 31;
  int lrow = lane & 15;                           // A row / B col owned by this lane
  int half = lane >> 4;                           // K-half select per ISA layout

  const uint4* xr0 = (const uint4*)(X  + (size_t)(mg * 32 +      lrow) * Kpad);
  const uint4* xr1 = (const uint4*)(X  + (size_t)(mg * 32 + 16 + lrow) * Kpad);
  const uint4* wr0 = (const uint4*)(Wt + (size_t)(ng * 32 +      lrow) * Kpad);
  const uint4* wr1 = (const uint4*)(Wt + (size_t)(ng * 32 + 16 + lrow) * Kpad);

  v8f acc00 = {}, acc01 = {}, acc10 = {}, acc11 = {};
  for (int kt = 0; kt < KT; ++kt) {
    FragU a0, a1, b0, b1;
    // 16-bit 16x32 fragment: lanes 0-15 hold K{0..7,16..23}, lanes 16-31 hold K{8..15,24..31}
    a0.u[0] = xr0[kt * 4 + half];  a0.u[1] = xr0[kt * 4 + 2 + half];
    a1.u[0] = xr1[kt * 4 + half];  a1.u[1] = xr1[kt * 4 + 2 + half];
    b0.u[0] = wr0[kt * 4 + half];  b0.u[1] = wr0[kt * 4 + 2 + half];
    b1.u[0] = wr1[kt * 4 + half];  b1.u[1] = wr1[kt * 4 + 2 + half];
    acc00 = __builtin_amdgcn_wmma_f32_16x16x32_bf16(false, a0.v, false, b0.v, (short)0, acc00, false, false);
    acc01 = __builtin_amdgcn_wmma_f32_16x16x32_bf16(false, a0.v, false, b1.v, (short)0, acc01, false, false);
    acc10 = __builtin_amdgcn_wmma_f32_16x16x32_bf16(false, a1.v, false, b0.v, (short)0, acc10, false, false);
    acc11 = __builtin_amdgcn_wmma_f32_16x16x32_bf16(false, a1.v, false, b1.v, (short)0, acc11, false, false);
  }

  #pragma unroll
  for (int ni = 0; ni < 2; ++ni) {
    int n = ng * 32 + ni * 16 + lrow;
    float scale = 0.f, shift = 0.f;
    if (n < Nvalid) {
      scale = gam[n] * rsqrtf(rvar[n] + BN_EPS);
      shift = bet[n] + (bias[n] - rmean[n]) * scale;  // z = acc*scale + shift
    }
    #pragma unroll
    for (int mi = 0; mi < 2; ++mi) {
      const v8f& acc = (mi == 0) ? (ni == 0 ? acc00 : acc01)
                                 : (ni == 0 ? acc10 : acc11);
      #pragma unroll
      for (int r = 0; r < 8; ++r) {               // D layout: m = r + 8*half, n = lane&15
        int m = mg * 32 + mi * 16 + r + half * 8;
        float z = acc[r] * scale + shift;
        z = fmaxf(z, 0.f);
        Y[(size_t)m * ldY + n] = f2bf(z);
      }
    }
  }
}

// ---------------- final: out = sigmoid(secsum + h3 @ Wo + bo) ----------------
__global__ __launch_bounds__(256) void k_final(const unsigned short* __restrict__ h3,
                                               const float* __restrict__ secsum,
                                               const float* __restrict__ Wo,
                                               const float* __restrict__ bo,
                                               float* __restrict__ out) {
  int wid = threadIdx.x >> 5, lane = threadIdx.x & 31;
  int b = blockIdx.x * 8 + wid;
  const unsigned short* hb = h3 + (size_t)b * HP;
  float s = 0.f;
  for (int n = lane; n < HID; n += 32) s += bf2f(hb[n]) * Wo[n];
  for (int m = 16; m; m >>= 1) s += __shfl_xor(s, m, 32);
  if (lane == 0) {
    float v = s + secsum[b] + bo[0];
    out[b] = 1.f / (1.f + expf(-v));
  }
}

// ---------------- host launcher ----------------
extern "C" void kernel_launch(void* const* d_in, const int* in_sizes, int n_in,
                              void* d_out, int out_size, void* d_ws, size_t ws_size,
                              hipStream_t stream) {
  const long long* x   = (const long long*)d_in[0];
  const float* emb     = (const float*)d_in[1];
  const float* cov     = (const float*)d_in[2];
  const float* W1 = (const float*)d_in[3];
  const float* b1 = (const float*)d_in[4];
  const float* g1 = (const float*)d_in[5];
  const float* be1 = (const float*)d_in[6];
  const float* rm1 = (const float*)d_in[7];
  const float* rv1 = (const float*)d_in[8];
  const float* W2 = (const float*)d_in[9];
  const float* b2 = (const float*)d_in[10];
  const float* g2 = (const float*)d_in[11];
  const float* be2 = (const float*)d_in[12];
  const float* rm2 = (const float*)d_in[13];
  const float* rv2 = (const float*)d_in[14];
  const float* W3 = (const float*)d_in[15];
  const float* b3 = (const float*)d_in[16];
  const float* g3 = (const float*)d_in[17];
  const float* be3 = (const float*)d_in[18];
  const float* rm3 = (const float*)d_in[19];
  const float* rv3 = (const float*)d_in[20];
  const float* Wo = (const float*)d_in[21];
  const float* bo = (const float*)d_in[22];
  float* out = (float*)d_out;

  // workspace carve-up (256B aligned)
  char* ws = (char*)d_ws;
  size_t o = 0;
  auto carve = [&](size_t bytes) { char* p = ws + o; o = (o + bytes + 255) & ~(size_t)255; return p; };
  unsigned short* h0  = (unsigned short*)carve((size_t)BATCH * K0P * 2);
  unsigned short* h1  = (unsigned short*)carve((size_t)BATCH * HP * 2);
  unsigned short* h2  = (unsigned short*)carve((size_t)BATCH * HP * 2);
  unsigned short* h3  = (unsigned short*)carve((size_t)BATCH * HP * 2);
  unsigned short* Wt1 = (unsigned short*)carve((size_t)HP * K0P * 2);
  unsigned short* Wt2 = (unsigned short*)carve((size_t)HP * HP * 2);
  unsigned short* Wt3 = (unsigned short*)carve((size_t)HP * HP * 2);
  float*          ss  = (float*)carve((size_t)BATCH * 4);

  // 1) weight convert + transpose to bf16 (N padded to HP with zero rows)
  {
    int t1 = HP * K0P, t2 = HP * HP;
    k_convert_wt<<<(t1 + 255) / 256, 256, 0, stream>>>(W1, Wt1, IN0, HID, K0P, HP);
    k_convert_wt<<<(t2 + 255) / 256, 256, 0, stream>>>(W2, Wt2, HID, HID, HP, HP);
    k_convert_wt<<<(t2 + 255) / 256, 256, 0, stream>>>(W3, Wt3, HID, HID, HP, HP);
  }
  // 2) gather embeddings + FM second order (8 waves/block, 1 row/wave)
  k_gather_fm<<<BATCH / 8, 256, 0, stream>>>(x, emb, cov, h0, ss);

  // 3) MLP layers via bf16 WMMA, 32x32 tile per wave
  int mgroups = (BATCH / 16) / 2;                      // 128
  int ngroups = (HP / 16) / 2;                         // 13 (covers all 416 cols; >=400 zeroed)
  int tasks = mgroups * ngroups;                       // 1664
  int blocks = (tasks + 7) / 8;                        // 8 waves per block
  k_gemm_bn_relu<<<blocks, 256, 0, stream>>>(h0, Wt1, h1, b1, g1, be1, rm1, rv1,
                                             mgroups, ngroups, K0P / 32, K0P, HP, HID);
  k_gemm_bn_relu<<<blocks, 256, 0, stream>>>(h1, Wt2, h2, b2, g2, be2, rm2, rv2,
                                             mgroups, ngroups, HP / 32, HP, HP, HID);
  k_gemm_bn_relu<<<blocks, 256, 0, stream>>>(h2, Wt3, h3, b3, g3, be3, rm3, rv3,
                                             mgroups, ngroups, HP / 32, HP, HP, HID);
  // 4) output head
  k_final<<<BATCH / 8, 256, 0, stream>>>(h3, ss, Wo, bo, out);
}
